// LSNNPolicy_22033182228694
// MI455X (gfx1250) — matrix operand order repaired and checked by reading, not compiled, not run on hardware
//
#include <hip/hip_runtime.h>
#include <hip/hip_bf16.h>

typedef _Float16 v16h __attribute__((ext_vector_type(16)));
typedef _Float16 v8h  __attribute__((ext_vector_type(8)));
typedef float    v8f  __attribute__((ext_vector_type(8)));

#define NB    16384   // batch
#define NH    128     // hidden
#define NT    40      // time steps
#define BT    64      // batch rows per workgroup
#define NTHR  512     // 16 waves (wave32)
#define ZSTR  168     // z-buffer row stride in halves: 128 z + 8 enc + 24 zero + 8 pad (bank-friendly)
#define ZDSTR 136     // dropped-spike buffer row stride (bank-friendly)
#define WOSTR 136     // w_out staging row stride

// LIF / LSNN constants (fp32, matching reference)
#define C_VM   0.1f                       // DT*TAU_MEM_INV
#define C_SYN  0.2f                       // DT*TAU_SYN_INV
#define C_ADPT 1.4285714285714286e-6f     // DT*TAU_ADAPT_INV
#define C_BETA 2.5714285714285714e-3f     // TAU_ADAPT_INV*BETA

union HU { v16h v; v8h h[2]; };

// A-fragment (16x32 f16): lane l holds 16 contiguous K at (k0 + 16*hsel) of row (l%16)
__device__ __forceinline__ v16h ldsA(const _Float16* row, int k0, int hsel) {
  HU u;
  u.h[0] = *(const v8h*)(row + k0 + 16 * hsel);
  u.h[1] = *(const v8h*)(row + k0 + 16 * hsel + 8);
  return u.v;
}
// B-fragment (32x16 f16): lane l holds 16 contiguous K at (k0 + 16*hsel) of column (l%16)
__device__ __forceinline__ v16h ldsB(const _Float16* p) {
  HU u;
  u.h[0] = *(const v8h*)(p);
  u.h[1] = *(const v8h*)(p + 8);
  return u.v;
}

#define WMMA16(a, bm, c) \
  __builtin_amdgcn_wmma_f32_16x16x32_f16(false, (a), false, (bm), (short)0, (c), false, false)

__global__ __launch_bounds__(NTHR, 1)
void lsnn_fused_kernel(const float* __restrict__ x,
                       const float* __restrict__ w_in,
                       const float* __restrict__ w_rec,
                       const float* __restrict__ w_out,
                       const float* __restrict__ drop,
                       float* __restrict__ out)
{
  __shared__ alignas(16) _Float16 zA[BT * ZSTR];       // spikes ping
  __shared__ alignas(16) _Float16 zB[BT * ZSTR];       // spikes pong
  __shared__ alignas(16) _Float16 zD[BT * ZDSTR];      // dropped spikes (z*mask)
  __shared__ alignas(16) _Float16 wo[2 * 16 * WOSTR];  // w_out hi/lo, padded to 16 cols
  __shared__ float vacc[BT * 2];                       // final max-voltages

  const int tid  = threadIdx.x;
  const int lane = tid & 31;
  const int wave = __builtin_amdgcn_readfirstlane(tid >> 5);  // uniform -> scalar branches
  const int hsel = lane >> 4;
  const int ln16 = lane & 15;
  const int mgrp = wave >> 3;     // 0..1 : which 32-row half
  const int nt   = wave & 7;      // 0..7 : hidden 16-col strip
  const int n0   = nt * 16;
  const int ncol = n0 + ln16;     // this lane's hidden column for state/B-frags
  const int wgrow = blockIdx.x * BT;

  // ---------------- init LDS ----------------
  {
    unsigned* a32 = (unsigned*)zA;
    unsigned* b32 = (unsigned*)zB;
    for (int idx = tid; idx < BT * ZSTR / 2; idx += NTHR) { a32[idx] = 0u; b32[idx] = 0u; }
    for (int idx = tid; idx < 2 * 16 * WOSTR; idx += NTHR) {
      int hl = idx / (16 * WOSTR);
      int rem = idx - hl * (16 * WOSTR);
      int n = rem / WOSTR;
      int k = rem - n * WOSTR;
      float w = (n < 2 && k < NH) ? w_out[n * NH + k] : 0.0f;
      _Float16 hi = (_Float16)w;
      wo[idx] = (hl == 0) ? hi : (_Float16)(w - (float)hi);
    }
  }

  // ---- cache recurrent B-fragments (w_rec | w_in | 0), hi/lo split, in registers ----
  v16h bhi[5], blo[5];
  {
#pragma unroll
    for (int kk = 0; kk < 5; ++kk) {
#pragma unroll
      for (int j = 0; j < 16; ++j) {
        int kg = kk * 32 + hsel * 16 + j;
        float w = 0.0f;
        if (kg < NH)            w = w_rec[ncol * NH + kg];
        else if (kg < NH + 8)   w = w_in[ncol * 8 + (kg - NH)];
        _Float16 hi = (_Float16)w;
        bhi[kk][j] = hi;
        blo[kk][j] = (_Float16)(w - (float)hi);
      }
    }
  }

  // ---------------- state (C-fragment layout) ----------------
  v8f   iacc[2];
  float v[2][8], b[2][8];
#pragma unroll
  for (int tt = 0; tt < 2; ++tt)
#pragma unroll
    for (int r = 0; r < 8; ++r) { iacc[tt][r] = 0.0f; v[tt][r] = 0.0f; b[tt][r] = 0.0f; }

  // encoder state: waves 0-1 own one row each (tid 0..63), 8 encoder neurons per row
  float ve[8], cur[8];
  if (wave < 2) {
    int row = tid;
#pragma unroll
    for (int k = 0; k < 4; ++k) {
      float xv = 50.0f * x[(size_t)(wgrow + row) * 4 + k];
      cur[k]     = xv > 0.0f ? xv : 0.0f;
      cur[k + 4] = -xv > 0.0f ? -xv : 0.0f;
    }
#pragma unroll
    for (int k = 0; k < 8; ++k) ve[k] = 0.0f;
  }

  // readout state: waves 0-3, D-fragment layout over [64 x 16] (cols >=2 are zero-padded)
  float vo[8], io[8], vmx[8];
#pragma unroll
  for (int r = 0; r < 8; ++r) { vo[r] = 0.0f; io[r] = 0.0f; vmx[r] = -3.4e38f; }

  // prefetch dropout mask for t=0
  float mpre[2][8];
#pragma unroll
  for (int tt = 0; tt < 2; ++tt) {
    int m0 = (mgrp * 2 + tt) * 16;
#pragma unroll
    for (int r = 0; r < 8; ++r) {
      int m = m0 + r + 8 * hsel;
      mpre[tt][r] = drop[((size_t)0 * NB + wgrow + m) * NH + ncol];
    }
  }

  __syncthreads();

  // ---------------- time loop ----------------
#pragma unroll 1
  for (int t = 0; t < NT; ++t) {
    _Float16* zr = (t & 1) ? zB : zA;   // holds z_{t-1}; tail gets ze_t
    _Float16* zw = (t & 1) ? zA : zB;   // receives z_t

    // ---- phase E: elementwise neuron update, emit spikes ----
#pragma unroll
    for (int tt = 0; tt < 2; ++tt) {
      int m0 = (mgrp * 2 + tt) * 16;
#pragma unroll
      for (int r = 0; r < 8; ++r) {
        int m = m0 + r + 8 * hsel;
        float vv = v[tt][r], ii = iacc[tt][r], bb = b[tt][r];
        float vd = vv + C_VM * (ii - vv);            // v_dec
        float bd = bb + C_ADPT * (1.0f - bb);        // b_dec
        float z  = (vd - bd) > 0.0f ? 1.0f : 0.0f;   // superspike fwd
        v[tt][r]    = (1.0f - z) * vd;
        b[tt][r]    = bd + z * C_BETA;
        iacc[tt][r] = ii - C_SYN * ii;               // i_dec (accumulator pre-scale)
        zw[m * ZSTR + ncol]  = (_Float16)z;
        zD[m * ZDSTR + ncol] = (_Float16)(z * mpre[tt][r]);
      }
    }

    // encoder spikes for THIS step into the tail of the read buffer
    if (wave < 2) {
      int row = tid;
#pragma unroll
      for (int k = 0; k < 8; ++k) {
        float vv = ve[k];
        vv = vv + C_VM * (cur[k] - vv);
        float z = (vv - 1.0f) > 0.0f ? 1.0f : 0.0f;
        ve[k] = (1.0f - z) * vv;
        zr[row * ZSTR + 128 + k] = (_Float16)z;
      }
    }

    // prefetch next step's dropout mask (hidden behind the WMMA phase)
    if (t + 1 < NT) {
#pragma unroll
      for (int tt = 0; tt < 2; ++tt) {
        int m0 = (mgrp * 2 + tt) * 16;
#pragma unroll
        for (int r = 0; r < 8; ++r) {
          int m = m0 + r + 8 * hsel;
          mpre[tt][r] = drop[((size_t)(t + 1) * NB + wgrow + m) * NH + ncol];
        }
      }
    }

    __syncthreads();

    // ---- phase M: i_new = i_dec + [z_{t-1} | ze_t] @ [w_rec | w_in]^T  (hi/lo split) ----
#pragma unroll
    for (int tt = 0; tt < 2; ++tt) {
      int m0 = (mgrp * 2 + tt) * 16;
      const _Float16* arow = zr + (m0 + ln16) * ZSTR;
      v8f c = iacc[tt];
#pragma unroll
      for (int kk = 0; kk < 5; ++kk) {
        v16h a = ldsA(arow, kk * 32, hsel);
        c = WMMA16(a, bhi[kk], c);
        c = WMMA16(a, blo[kk], c);
      }
      iacc[tt] = c;
    }

    // ---- readout: (z_t * mask) @ w_out^T on waves 0-3, then LI cell update ----
    if (wave < 4) {
      int m0r = wave * 16;
      const _Float16* arow = zD + (m0r + ln16) * ZDSTR;
      v8f d;
#pragma unroll
      for (int r = 0; r < 8; ++r) d[r] = 0.0f;
#pragma unroll
      for (int kk = 0; kk < 4; ++kk) {
        v16h a  = ldsA(arow, kk * 32, hsel);
        v16h wh = ldsB(&wo[(0 * 16 + ln16) * WOSTR + kk * 32 + 16 * hsel]);
        v16h wl = ldsB(&wo[(1 * 16 + ln16) * WOSTR + kk * 32 + 16 * hsel]);
        d = WMMA16(a, wh, d);
        d = WMMA16(a, wl, d);
      }
#pragma unroll
      for (int r = 0; r < 8; ++r) {
        float von = vo[r] + C_VM * (io[r] - vo[r]);   // vo_new uses old io
        io[r] = io[r] - C_SYN * io[r] + d[r];
        vo[r] = von;
        vmx[r] = fmaxf(vmx[r], von);
      }
    }

    __syncthreads();
  }

  // ---------------- epilogue: gather vmax, softmax over O=2 ----------------
  if (wave < 4) {
    if (ln16 < 2) {
      int m0r = wave * 16;
#pragma unroll
      for (int r = 0; r < 8; ++r) {
        int m = m0r + r + 8 * hsel;
        vacc[m * 2 + ln16] = vmx[r];
      }
    }
  }
  __syncthreads();

  if (tid < BT) {
    float v0 = vacc[tid * 2 + 0];
    float v1 = vacc[tid * 2 + 1];
    float mx = fmaxf(v0, v1);
    float e0 = expf(v0 - mx);
    float e1 = expf(v1 - mx);
    float s  = e0 + e1;
    out[(size_t)(wgrow + tid) * 2 + 0] = e0 / s;
    out[(size_t)(wgrow + tid) * 2 + 1] = e1 / s;
  }
}

extern "C" void kernel_launch(void* const* d_in, const int* in_sizes, int n_in,
                              void* d_out, int out_size, void* d_ws, size_t ws_size,
                              hipStream_t stream) {
  (void)in_sizes; (void)n_in; (void)out_size; (void)d_ws; (void)ws_size;
  const float* x     = (const float*)d_in[0];
  const float* w_in  = (const float*)d_in[1];
  const float* w_rec = (const float*)d_in[2];
  const float* w_out = (const float*)d_in[3];
  const float* drop  = (const float*)d_in[4];
  float* out = (float*)d_out;
  hipLaunchKernelGGL(lsnn_fused_kernel, dim3(NB / BT), dim3(NTHR), 0, stream,
                     x, w_in, w_rec, w_out, drop, out);
}
